// ArchitectModule_74036646248739
// MI455X (gfx1250) — matrix operand order repaired and checked by reading, not compiled, not run on hardware
//
#include <hip/hip_runtime.h>

typedef __attribute__((ext_vector_type(16))) __bf16 v16bf;
typedef __attribute__((ext_vector_type(8)))  float  v8f;

namespace {
constexpr int D        = 1024;
constexpr int CTX      = 3072;
constexpr int B        = 512;
constexpr int T        = 64;
constexpr int RPW      = 16;                // rows per workgroup (one WMMA M tile)
constexpr int NWG      = B / RPW;           // 32
constexpr int NWAVES   = 16;
constexpr int NTHREADS = NWAVES * 32;       // 512
constexpr int NT       = (D / 16) / NWAVES; // 4 column tiles per wave
constexpr int KT_D     = D / 32;            // 32 k-tiles for K=1024
constexpr int KT_CTX   = CTX / 32;          // 96 k-tiles for K=3072
constexpr size_t W_PACK_BYTES = (size_t)D * 3 * D * 2;   // 6 MiB per packed weight
}

// byte address of element (m,kk) inside a 1024B 16x32 bf16 WMMA A tile
__device__ __forceinline__ int a_frag_byte(int m, int kk) {
  int group = kk >> 3;            // 0..3 (K 0-7 / 8-15 / 16-23 / 24-31)
  int lo    = kk & 7;
  int lane  = m + ((group & 1) << 4);
  int off   = ((group >> 1) << 4) + ((lo >> 1) << 2) + ((lo & 1) << 1);
  return lane * 32 + off;
}

// byte address of element (kk,nn) inside a 1024B 32x16 bf16 WMMA B tile
__device__ __forceinline__ int b_frag_byte(int kk, int nn) {
  int lane = nn + (((kk >> 3) & 1) << 4);
  return lane * 32 + ((((kk >> 4) << 3) + (kk & 7)) << 1);
}

__device__ __forceinline__ float sigm(float x) { return 1.f / (1.f + __expf(-x)); }
__device__ __forceinline__ float tanh_f(float x) {
  float e = __expf(2.f * x);
  return 1.f - 2.f / (e + 1.f);
}

__device__ __forceinline__ v8f wmma_bf16(v16bf a, v16bf b, v8f c) {
  return __builtin_amdgcn_wmma_f32_16x16x32_bf16(false, a, false, b, (short)0, c,
                                                 false, false);
}

// ---------------- pre-pass: pack f32 weight [K][N] into bf16 B-fragment tiles
// layout: tile(jt,kt) at ((jt*(K/32))+kt)*1024 bytes, jt = n>>4
__global__ void pack_weight(const float* __restrict__ src, __bf16* __restrict__ dst,
                            int K, int N) {
  int total = K * N;
  for (int i = blockIdx.x * blockDim.x + threadIdx.x; i < total;
       i += gridDim.x * blockDim.x) {
    int k = i / N, n = i - k * N;
    int jt = n >> 4, nn = n & 15, kt = k >> 5, kk = k & 31;
    int byteaddr = (jt * (K >> 5) + kt) * 1024 + b_frag_byte(kk, nn);
    dst[byteaddr >> 1] = (__bf16)src[i];
  }
}

// ---------------- pre-pass: gi0 = b_ih + start_token @ W_ih   (row vector, [3D])
__global__ void gi0_kernel(const float* __restrict__ st, const float* __restrict__ Wih,
                           const float* __restrict__ bih, float* __restrict__ gi0) {
  int n = blockIdx.x * blockDim.x + threadIdx.x;
  if (n >= 3 * D) return;
  float acc = bih[n];
  for (int k = 0; k < D; ++k) acc = fmaf(st[k], Wih[k * (3 * D) + n], acc);
  gi0[n] = acc;
}

// ---------------- h0 = [ws|goal] @ W_ctx + b_ctx  (bf16 WMMA, f32 out)
__global__ __launch_bounds__(NTHREADS) void h0_kernel(
    const float* __restrict__ wstate, const float* __restrict__ goal,
    const __bf16* __restrict__ Wctx_p, const float* __restrict__ bctx,
    float* __restrict__ h0) {
  __shared__ __align__(32) char ldsA[32 * 1024];
  const int tid = threadIdx.x, wg = blockIdx.x;
  const int lane = tid & 31, wave = tid >> 5;
  const int row0 = wg * RPW;
  const int mb = (lane >> 4) << 3;

  v8f acc[NT] = {};
  for (int chunk = 0; chunk < 3; ++chunk) {
    __syncthreads();
    // stage 16 rows x 1024 K of ctx as bf16 in A-fragment layout
    for (int i = tid; i < RPW * 1024; i += NTHREADS) {
      int m = i >> 10, kloc = i & 1023;
      int kg = chunk * 1024 + kloc;
      float v = (kg < 2048) ? wstate[(size_t)(row0 + m) * 2048 + kg]
                            : goal[(size_t)(row0 + m) * 1024 + (kg - 2048)];
      *(__bf16*)(ldsA + ((kloc >> 5) << 10) + a_frag_byte(m, kloc & 31)) = (__bf16)v;
    }
    __syncthreads();
    for (int jj = 0; jj < NT; ++jj) {
      int j = wave * NT + jj;
      const v16bf* bp =
          (const v16bf*)Wctx_p + (size_t)(j * KT_CTX + chunk * 32) * 32 + lane;
      #pragma unroll 4
      for (int kt = 0; kt < 32; ++kt) {
        v16bf a = *(const v16bf*)(ldsA + (kt << 10) + lane * 32);
        acc[jj] = wmma_bf16(a, bp[kt * 32], acc[jj]);
      }
    }
  }
  for (int jj = 0; jj < NT; ++jj) {
    int col = (wave * NT + jj) * 16 + (lane & 15);
    float bc = bctx[col];
    #pragma unroll
    for (int r = 0; r < 8; ++r)
      h0[(size_t)(row0 + mb + r) * D + col] = acc[jj][r] + bc;
  }
}

// stage one wave's h tiles (f32 regs, WMMA C layout) into an LDS A-fragment buffer
__device__ __forceinline__ void stage_h(char* buf, const float h[][8],
                                        const int* col, int mb) {
  #pragma unroll
  for (int jj = 0; jj < NT; ++jj) {
    int c = col[jj], kt = c >> 5, kk = c & 31;
    #pragma unroll
    for (int r = 0; r < 8; ++r)
      *(__bf16*)(buf + (kt << 10) + a_frag_byte(mb + r, kk)) = (__bf16)h[jj][r];
  }
}

// ---------------- recurrent GRU: h lives in regs (f32) + double-buffered LDS
__global__ __launch_bounds__(NTHREADS) void gru_kernel(
    const __bf16* __restrict__ Wih_p, const __bf16* __restrict__ Whh_p,
    const float* __restrict__ bih, const float* __restrict__ bhh,
    const float* __restrict__ gi0, const float* __restrict__ h0,
    float* __restrict__ out) {
  __shared__ __align__(32) char ldsH[2][KT_D * 1024];   // 2 x 32 KB
  const int tid = threadIdx.x, wg = blockIdx.x;
  const int lane = tid & 31, wave = tid >> 5;
  const int row0 = wg * RPW;
  const int mb = (lane >> 4) << 3;

  float h[NT][8];
  float c_bir[NT], c_biz[NT], c_bin[NT], c_bhr[NT], c_bhz[NT], c_bhn[NT];
  float c_g0r[NT], c_g0z[NT], c_g0n[NT];
  int   col[NT];

  #pragma unroll
  for (int jj = 0; jj < NT; ++jj) {
    int c = (wave * NT + jj) * 16 + (lane & 15);
    col[jj] = c;
    c_bir[jj] = bih[c];           c_bhr[jj] = bhh[c];
    c_biz[jj] = bih[c + D];       c_bhz[jj] = bhh[c + D];
    c_bin[jj] = bih[c + 2 * D];   c_bhn[jj] = bhh[c + 2 * D];
    c_g0r[jj] = gi0[c]; c_g0z[jj] = gi0[c + D]; c_g0n[jj] = gi0[c + 2 * D];
    #pragma unroll
    for (int r = 0; r < 8; ++r) h[jj][r] = h0[(size_t)(row0 + mb + r) * D + c];
  }
  stage_h(ldsH[0], h, col, mb);
  __syncthreads();

  // ---- step 0: gi from precomputed broadcast gi0; read buf 0, write buf 1 ----
  for (int jj = 0; jj < NT; ++jj) {
    int j = wave * NT + jj;
    v8f ar = {}, az = {}, an = {};
    const v16bf* br = (const v16bf*)Whh_p + (size_t)((0 * 64 + j) * KT_D) * 32 + lane;
    const v16bf* bz = (const v16bf*)Whh_p + (size_t)((1 * 64 + j) * KT_D) * 32 + lane;
    const v16bf* bn = (const v16bf*)Whh_p + (size_t)((2 * 64 + j) * KT_D) * 32 + lane;
    #pragma unroll 4
    for (int kt = 0; kt < KT_D; ++kt) {
      v16bf a = *(const v16bf*)(ldsH[0] + (kt << 10) + lane * 32);
      ar = wmma_bf16(a, br[kt * 32], ar);
      az = wmma_bf16(a, bz[kt * 32], az);
      an = wmma_bf16(a, bn[kt * 32], an);
    }
    #pragma unroll
    for (int r = 0; r < 8; ++r) {
      float rr = sigm(c_g0r[jj] + ar[r] + c_bhr[jj]);
      float zz = sigm(c_g0z[jj] + az[r] + c_bhz[jj]);
      float nn = tanh_f(c_g0n[jj] + rr * (an[r] + c_bhn[jj]));
      float hn = (1.f - zz) * nn + zz * h[jj][r];
      h[jj][r] = hn;
      __builtin_nontemporal_store(
          hn, &out[((size_t)(row0 + mb + r) * T + 0) * D + col[jj]]);
    }
  }
  stage_h(ldsH[1], h, col, mb);
  __syncthreads();

  // ---- steps 1..T-1: x == h; read buf t&1, write buf (t+1)&1, 1 barrier/step ----
  for (int t = 1; t < T; ++t) {
    char* rbuf = ldsH[t & 1];
    for (int jj = 0; jj < NT; ++jj) {
      int j = wave * NT + jj;
      v8f air = {}, aiz = {}, ain = {}, ahr = {}, ahz = {}, ahn = {};
      const v16bf* bir = (const v16bf*)Wih_p + (size_t)((0 * 64 + j) * KT_D) * 32 + lane;
      const v16bf* biz = (const v16bf*)Wih_p + (size_t)((1 * 64 + j) * KT_D) * 32 + lane;
      const v16bf* bin = (const v16bf*)Wih_p + (size_t)((2 * 64 + j) * KT_D) * 32 + lane;
      const v16bf* bhr = (const v16bf*)Whh_p + (size_t)((0 * 64 + j) * KT_D) * 32 + lane;
      const v16bf* bhz = (const v16bf*)Whh_p + (size_t)((1 * 64 + j) * KT_D) * 32 + lane;
      const v16bf* bhn = (const v16bf*)Whh_p + (size_t)((2 * 64 + j) * KT_D) * 32 + lane;
      #pragma unroll 4
      for (int kt = 0; kt < KT_D; ++kt) {
        v16bf a = *(const v16bf*)(rbuf + (kt << 10) + lane * 32);
        air = wmma_bf16(a, bir[kt * 32], air);
        aiz = wmma_bf16(a, biz[kt * 32], aiz);
        ain = wmma_bf16(a, bin[kt * 32], ain);
        ahr = wmma_bf16(a, bhr[kt * 32], ahr);
        ahz = wmma_bf16(a, bhz[kt * 32], ahz);
        ahn = wmma_bf16(a, bhn[kt * 32], ahn);
      }
      #pragma unroll
      for (int r = 0; r < 8; ++r) {
        float rr = sigm(air[r] + ahr[r] + c_bir[jj] + c_bhr[jj]);
        float zz = sigm(aiz[r] + ahz[r] + c_biz[jj] + c_bhz[jj]);
        float nn = tanh_f(ain[r] + c_bin[jj] + rr * (ahn[r] + c_bhn[jj]));
        float hn = (1.f - zz) * nn + zz * h[jj][r];
        h[jj][r] = hn;
        __builtin_nontemporal_store(
            hn, &out[((size_t)(row0 + mb + r) * T + t) * D + col[jj]]);
      }
    }
    stage_h(ldsH[(t + 1) & 1], h, col, mb);
    __syncthreads();
  }
}

extern "C" void kernel_launch(void* const* d_in, const int* in_sizes, int n_in,
                              void* d_out, int out_size, void* d_ws, size_t ws_size,
                              hipStream_t stream) {
  const float* world = (const float*)d_in[0];
  const float* goal  = (const float*)d_in[1];
  const float* Wctx  = (const float*)d_in[2];
  const float* bctx  = (const float*)d_in[3];
  const float* start = (const float*)d_in[4];
  const float* Wih   = (const float*)d_in[5];
  const float* bih   = (const float*)d_in[6];
  const float* Whh   = (const float*)d_in[7];
  const float* bhh   = (const float*)d_in[8];
  float* out = (float*)d_out;

  char* ws = (char*)d_ws;
  __bf16* Wih_p  = (__bf16*)(ws);
  __bf16* Whh_p  = (__bf16*)(ws + W_PACK_BYTES);
  __bf16* Wctx_p = (__bf16*)(ws + 2 * W_PACK_BYTES);
  float*  h0     = (float*)(ws + 3 * W_PACK_BYTES);
  float*  gi0    = (float*)(ws + 3 * W_PACK_BYTES + (size_t)B * D * 4);

  pack_weight<<<2048, 256, 0, stream>>>(Wih, Wih_p, D, 3 * D);
  pack_weight<<<2048, 256, 0, stream>>>(Whh, Whh_p, D, 3 * D);
  pack_weight<<<2048, 256, 0, stream>>>(Wctx, Wctx_p, CTX, D);
  gi0_kernel<<<(3 * D + 255) / 256, 256, 0, stream>>>(start, Wih, bih, gi0);
  h0_kernel<<<NWG, NTHREADS, 0, stream>>>(world, goal, Wctx_p, bctx, h0);
  gru_kernel<<<NWG, NTHREADS, 0, stream>>>(Wih_p, Whh_p, bih, bhh, gi0, h0, out);
}
